// AttFlat_2757369004817
// MI455X (gfx1250) — compile-verified, matrix operand order checked by reading, not run on hardware
//
#include <hip/hip_runtime.h>
#include <math.h>

// ---------------------------------------------------------------------------
// Types for WMMA
// ---------------------------------------------------------------------------
typedef __bf16  bf16_t;
typedef __attribute__((ext_vector_type(16))) __bf16 v16bf;
typedef __attribute__((ext_vector_type(8)))  float  v8f;
typedef __attribute__((ext_vector_type(2)))  float  v2f;
typedef __attribute__((ext_vector_type(4)))  unsigned int v4u;

union FragBF {            // 16 bf16 = 32 bytes, loaded as two 16B chunks
    v16bf v;
    struct { v4u lo; v4u hi; } q;
};

// sizes
#define BATCH   512
#define SEQ     196
#define HIDDEN  1024
#define MID     512
#define NOUT    2048
#define NBASIS  100
#define NTOK    (BATCH*SEQ)          // 100352, divisible by 64

// workspace offsets (bytes)
#define OFF_W1BF   0                                  // 512*1024*2 = 1048576
#define OFF_LOGITS (OFF_W1BF + MID*HIDDEN*2)          // NTOK*4
#define OFF_WBUF   (OFF_LOGITS + NTOK*4)              // BATCH*SEQ*4
#define OFF_CTX    (OFF_WBUF + BATCH*SEQ*4)           // BATCH*HIDDEN*4

__device__ __forceinline__ unsigned short f2bf(float x) {
    unsigned int u = __float_as_uint(x);
    u += 0x7FFFu + ((u >> 16) & 1u);          // round-to-nearest-even
    return (unsigned short)(u >> 16);
}

// ---------------------------------------------------------------------------
// Kernel 0: convert W1 (f32 512x1024) -> bf16 workspace
// ---------------------------------------------------------------------------
__global__ void k_convert_w1(const float* __restrict__ W1,
                             unsigned short* __restrict__ w1bf) {
    int id = (blockIdx.x * 256 + threadIdx.x) * 4;
    if (id < MID * HIDDEN) {
        float4 v = *reinterpret_cast<const float4*>(W1 + id);
        unsigned int lo = (unsigned int)f2bf(v.x) | ((unsigned int)f2bf(v.y) << 16);
        unsigned int hi = (unsigned int)f2bf(v.z) | ((unsigned int)f2bf(v.w) << 16);
        *reinterpret_cast<uint2*>(w1bf + id) = make_uint2(lo, hi);
    }
}

// ---------------------------------------------------------------------------
// Kernel 1: fused  h = relu(x@W1^T + b1) ;  logit = h@W2 (+b2)
// Block = 64 tokens x all 512 mid columns (8 waves, each 64M x 64N).
// x staged to LDS as bf16 in K-chunks of 256; W1 (bf16) streamed from L2.
// bf16 WMMA 16x16x32, f32 accumulate.  h never materialized.
// ---------------------------------------------------------------------------
#define MTILE    64
#define KCHUNK   256
#define A_STRIDE 264   // halves per LDS row (264*2=528B) -> conflict-free b128

__global__ void __launch_bounds__(256)
k_mlp_logits(const float* __restrict__ x,
             const unsigned short* __restrict__ w1bf,
             const float* __restrict__ b1,
             const float* __restrict__ W2,
             const float* __restrict__ b2,
             float* __restrict__ logits) {
    __shared__ unsigned short Asm[MTILE * A_STRIDE];    // 33.8 KB
    __shared__ float part_s[8][MTILE];                  // 2 KB

    const int tid  = threadIdx.x;
    const int blk  = blockIdx.x;
    const int wave = tid >> 5;
    const int lane = tid & 31;
    const int nrow = lane & 15;
    const int rsel = lane >> 4;

    v8f acc[4][4];                                      // [Mtile][Ntile]
    #pragma unroll
    for (int mt = 0; mt < 4; ++mt)
        #pragma unroll
        for (int nt = 0; nt < 4; ++nt) acc[mt][nt] = (v8f)0.0f;

    const int nbase0 = wave * 64;

    for (int kc = 0; kc < HIDDEN; kc += KCHUNK) {
        if (kc) __syncthreads();                        // protect previous chunk
        // ---- stage 64 tokens x 256 K of x into LDS as bf16 ----
        const float* xt = x + (size_t)blk * (MTILE * HIDDEN) + kc;
        #pragma unroll
        for (int i = 0; i < 16; ++i) {
            int e   = (i * 256 + tid) * 4;              // 0..16383
            int row = e >> 8;                           // 256 cols / row
            int col = e & 255;
            float4 v = *reinterpret_cast<const float4*>(xt + (size_t)row * HIDDEN + col);
            unsigned int lo = (unsigned int)f2bf(v.x) | ((unsigned int)f2bf(v.y) << 16);
            unsigned int hi = (unsigned int)f2bf(v.z) | ((unsigned int)f2bf(v.w) << 16);
            *reinterpret_cast<uint2*>(&Asm[row * A_STRIDE + col]) = make_uint2(lo, hi);
        }
        __syncthreads();

        // ---- 16 WMMAs per K-step: 4 A frags (LDS) x 4 B frags (global) ----
        for (int k = 0; k < KCHUNK; k += 32) {
            FragBF a[4];
            #pragma unroll
            for (int mt = 0; mt < 4; ++mt) {
                const unsigned short* arow = &Asm[(mt * 16 + nrow) * A_STRIDE];
                a[mt].q.lo = *reinterpret_cast<const v4u*>(arow + k + rsel * 8);
                a[mt].q.hi = *reinterpret_cast<const v4u*>(arow + k + 16 + rsel * 8);
            }
            #pragma unroll
            for (int nt = 0; nt < 4; ++nt) {
                const unsigned short* brow =
                    w1bf + (size_t)(nbase0 + nt * 16 + nrow) * HIDDEN + kc;
                FragBF bfr;
                bfr.q.lo = *reinterpret_cast<const v4u*>(brow + k + rsel * 8);
                bfr.q.hi = *reinterpret_cast<const v4u*>(brow + k + 16 + rsel * 8);
                #pragma unroll
                for (int mt = 0; mt < 4; ++mt) {
                    acc[mt][nt] = __builtin_amdgcn_wmma_f32_16x16x32_bf16(
                        false, a[mt].v, false, bfr.v, (short)0, acc[mt][nt], false, false);
                }
            }
        }
    }

    // ---- epilogue: relu(acc + b1) * W2, reduce over N ----
    float p[4][8];
    #pragma unroll
    for (int mt = 0; mt < 4; ++mt)
        #pragma unroll
        for (int r = 0; r < 8; ++r) p[mt][r] = 0.0f;

    #pragma unroll
    for (int nt = 0; nt < 4; ++nt) {
        int n = nbase0 + nt * 16 + nrow;
        float bb = b1[n];
        float ww = W2[n];
        #pragma unroll
        for (int mt = 0; mt < 4; ++mt) {
            #pragma unroll
            for (int r = 0; r < 8; ++r) {
                float h = acc[mt][nt][r] + bb;
                h = h > 0.0f ? h : 0.0f;
                p[mt][r] += h * ww;
            }
        }
    }
    // reduce across the 16 lanes sharing the same M rows
    #pragma unroll
    for (int off = 1; off < 16; off <<= 1) {
        #pragma unroll
        for (int mt = 0; mt < 4; ++mt)
            #pragma unroll
            for (int r = 0; r < 8; ++r) p[mt][r] += __shfl_xor(p[mt][r], off, 32);
    }
    if (lane == 0) {
        #pragma unroll
        for (int mt = 0; mt < 4; ++mt)
            #pragma unroll
            for (int r = 0; r < 8; ++r) part_s[wave][mt * 16 + r] = p[mt][r];
    }
    if (lane == 16) {
        #pragma unroll
        for (int mt = 0; mt < 4; ++mt)
            #pragma unroll
            for (int r = 0; r < 8; ++r) part_s[wave][mt * 16 + 8 + r] = p[mt][r];
    }
    __syncthreads();

    if (tid < MTILE) {   // deterministic fixed-order sum over waves
        float s = b2[0];
        #pragma unroll
        for (int w = 0; w < 8; ++w) s += part_s[w][tid];
        logits[blk * MTILE + tid] = s;
    }
}

// ---------------------------------------------------------------------------
// Kernel 2: per-batch softmax, 2x2 moment chain, Gaussian basis r, w = G r
// ---------------------------------------------------------------------------
__device__ __forceinline__ float breduce_sum(float v, volatile float* red, int tid) {
    red[tid] = v; __syncthreads();
    for (int s = 128; s > 0; s >>= 1) { if (tid < s) red[tid] += red[tid + s]; __syncthreads(); }
    float r = red[0]; __syncthreads();
    return r;
}
__device__ __forceinline__ float breduce_max(float v, volatile float* red, int tid) {
    red[tid] = v; __syncthreads();
    for (int s = 128; s > 0; s >>= 1) { if (tid < s) red[tid] = fmaxf(red[tid], red[tid + s]); __syncthreads(); }
    float r = red[0]; __syncthreads();
    return r;
}

__global__ void __launch_bounds__(256)
k_attention_stats(const float* __restrict__ logits,
                  const unsigned char* __restrict__ xmask,
                  const float* __restrict__ G,
                  const float* __restrict__ mu_basis,
                  float* __restrict__ wbuf) {
    __shared__ float red[256];
    __shared__ float sc[6];
    __shared__ float rbas[NBASIS];

    const int b   = blockIdx.x;
    const int tid = threadIdx.x;

    float lg = -1.0e30f;
    if (tid < SEQ) {
        lg = logits[b * SEQ + tid];
        if (xmask[b * SEQ + tid]) lg = -1.0e9f;
    }
    float mx = breduce_max(lg, red, tid);
    float e  = (tid < SEQ) ? expf(lg - mx) : 0.0f;
    float s  = breduce_sum(e, red, tid);

    float px = 0.0f, py = 0.0f;
    if (tid < SEQ) {
        int i = tid / 14, j = tid % 14;
        px = (float)i * (1.0f / 13.0f);
        py = (float)j * (1.0f / 13.0f);
    }
    float Sx  = breduce_sum(e * px,      red, tid);
    float Sy  = breduce_sum(e * py,      red, tid);
    float Sxx = breduce_sum(e * px * px, red, tid);
    float Sxy = breduce_sum(e * px * py, red, tid);
    float Syy = breduce_sum(e * py * py, red, tid);

    if (tid == 0) {
        float inv_s = 1.0f / s;
        float mux = Sx * inv_s, muy = Sy * inv_s;
        float a  = Sxx * inv_s - mux * mux + 1e-6f;
        float bb = Sxy * inv_s - mux * muy;
        float d  = Syy * inv_s - muy * muy + 1e-6f;
        float det = a * d - bb * bb;
        float i00 = d / det, i01 = -bb / det, i11 = a / det;
        float t0 = i00 * mux + i01 * muy;
        float t1 = i01 * mux + i11 * muy;
        float det2 = i00 * i11 - i01 * i01;
        float r00 = i11 / det2, r01 = -i01 / det2, r11 = i00 / det2;
        float m0 = r00 * t0 + r01 * t1;
        float m1 = r01 * t0 + r11 * t1;
        float a00 = r00 + 1e-3f, a01 = r01, a11 = r11 + 1e-3f;
        float detA = a00 * a11 - a01 * a01;
        float ai00 = a11 / detA, ai01 = -a01 / detA, ai11 = a00 / detA;
        float coef = 1.0f / (2.0f * 3.14159265358979323846f * sqrtf(detA));
        sc[0] = m0; sc[1] = m1; sc[2] = ai00; sc[3] = ai01; sc[4] = ai11; sc[5] = coef;
    }
    __syncthreads();

    if (tid < NBASIS) {
        float d0 = sc[0] - mu_basis[tid * 2 + 0];
        float d1 = sc[1] - mu_basis[tid * 2 + 1];
        float quad = sc[2] * d0 * d0 + 2.0f * sc[3] * d0 * d1 + sc[4] * d1 * d1;
        rbas[tid] = expf(-0.5f * quad) * sc[5];
    }
    __syncthreads();

    if (tid < SEQ) {
        float w = 0.0f;
        const float* g = G + tid * NBASIS;
        #pragma unroll 4
        for (int k = 0; k < NBASIS; ++k) w += g[k] * rbas[k];
        wbuf[b * SEQ + tid] = w;
    }
}

// ---------------------------------------------------------------------------
// Kernel 3: context[b][h] = sum_n wbuf[b][n] * x[b][n][h]
// ---------------------------------------------------------------------------
__global__ void __launch_bounds__(256)
k_context(const float* __restrict__ x,
          const float* __restrict__ wbuf,
          float* __restrict__ ctx) {
    __shared__ float ws[SEQ];
    const int b   = blockIdx.y;
    const int tid = threadIdx.x;
    const int h   = blockIdx.x * 256 + tid;
    if (tid < SEQ) ws[tid] = wbuf[b * SEQ + tid];
    __syncthreads();

    const float* xb = x + (size_t)b * SEQ * HIDDEN + h;
    float acc = 0.0f;
    #pragma unroll 4
    for (int n = 0; n < SEQ; ++n) acc += ws[n] * xb[(size_t)n * HIDDEN];
    ctx[b * HIDDEN + h] = acc;
}

// ---------------------------------------------------------------------------
// Kernel 4: out = ctx @ Wm^T + bm  in full f32 via WMMA 16x16x4.
// M=512, N=2048, K=1024.  8 waves/block, 4 N-tiles per wave.
// ---------------------------------------------------------------------------
__global__ void __launch_bounds__(256)
k_out_gemm(const float* __restrict__ ctx,
           const float* __restrict__ Wm,
           const float* __restrict__ bm,
           float* __restrict__ out) {
    const int tid  = threadIdx.x;
    const int wave = tid >> 5;
    const int lane = tid & 31;
    const int nrow = lane & 15;
    const int rsel = lane >> 4;

    const int mb    = blockIdx.y;                       // M tile (16 rows)
    const int ntile = blockIdx.x * 32 + wave * 4;       // first of 4 N tiles

    v8f acc[4];
    #pragma unroll
    for (int t = 0; t < 4; ++t) acc[t] = (v8f)0.0f;

    const float* arow = ctx + (size_t)(mb * 16 + nrow) * HIDDEN;

    for (int k = 0; k < HIDDEN; k += 4) {
        v2f a = *reinterpret_cast<const v2f*>(arow + k + 2 * rsel);
        #pragma unroll
        for (int t = 0; t < 4; ++t) {
            const float* brow = Wm + (size_t)((ntile + t) * 16 + nrow) * HIDDEN;
            v2f bfr = *reinterpret_cast<const v2f*>(brow + k + 2 * rsel);
            acc[t] = __builtin_amdgcn_wmma_f32_16x16x4_f32(
                false, a, false, bfr, (short)0, acc[t], false, false);
        }
    }

    #pragma unroll
    for (int t = 0; t < 4; ++t) {
        int n = (ntile + t) * 16 + nrow;
        float bias = bm[n];
        #pragma unroll
        for (int r = 0; r < 8; ++r) {
            int m = mb * 16 + r + 8 * rsel;
            out[(size_t)m * NOUT + n] = acc[t][r] + bias;
        }
    }
}

// ---------------------------------------------------------------------------
// Launch
// ---------------------------------------------------------------------------
extern "C" void kernel_launch(void* const* d_in, const int* in_sizes, int n_in,
                              void* d_out, int out_size, void* d_ws, size_t ws_size,
                              hipStream_t stream) {
    (void)in_sizes; (void)n_in; (void)out_size; (void)ws_size;

    const float*         x        = (const float*)d_in[0];
    const unsigned char* xmask    = (const unsigned char*)d_in[1];
    const float*         W1       = (const float*)d_in[2];
    const float*         b1       = (const float*)d_in[3];
    const float*         W2       = (const float*)d_in[4];
    const float*         b2       = (const float*)d_in[5];
    const float*         Wm       = (const float*)d_in[6];
    const float*         bm       = (const float*)d_in[7];
    const float*         G        = (const float*)d_in[8];
    const float*         mu_basis = (const float*)d_in[9];
    float*               out      = (float*)d_out;

    char* ws = (char*)d_ws;
    unsigned short* w1bf   = (unsigned short*)(ws + OFF_W1BF);
    float*          logits = (float*)(ws + OFF_LOGITS);
    float*          wbuf   = (float*)(ws + OFF_WBUF);
    float*          ctx    = (float*)(ws + OFF_CTX);

    k_convert_w1<<<(MID * HIDDEN) / (256 * 4), 256, 0, stream>>>(W1, w1bf);
    k_mlp_logits<<<NTOK / MTILE, 256, 0, stream>>>(x, w1bf, b1, W2, b2, logits);
    k_attention_stats<<<BATCH, 256, 0, stream>>>(logits, xmask, G, mu_basis, wbuf);
    k_context<<<dim3(HIDDEN / 256, BATCH), 256, 0, stream>>>(x, wbuf, ctx);
    k_out_gemm<<<dim3(4, BATCH / 16), 256, 0, stream>>>(ctx, Wm, bm, out);
}
